// GraphAttentionLayer_47777216201171
// MI455X (gfx1250) — matrix operand order repaired and checked by reading, not compiled, not run on hardware
//
#include <hip/hip_runtime.h>

// ---------------------------------------------------------------------------
// Problem constants (match reference)
// ---------------------------------------------------------------------------
constexpr int Bb  = 2;
constexpr int Nn  = 2048;
constexpr int Dd  = 256;
constexpr int Hh  = 8;
constexpr int DHh = Dd / Hh;          // 32
constexpr float EPSc = 1e-5f;

// ---------------------------------------------------------------------------
// WMMA vector types (CDNA5 wave32)
// ---------------------------------------------------------------------------
typedef __attribute__((ext_vector_type(16))) __bf16          v16bf;
typedef __attribute__((ext_vector_type(16))) unsigned short  v16us;
typedef __attribute__((ext_vector_type(8)))  unsigned short  ushort8;
typedef __attribute__((ext_vector_type(8)))  float           v8f;

// float -> bf16 round-to-nearest-even
__device__ __forceinline__ unsigned short f2bf(float f) {
    unsigned u = __float_as_uint(f);
    unsigned r = u + 0x7fffu + ((u >> 16) & 1u);
    return (unsigned short)(r >> 16);
}

// Load one A-fragment (16x32 bf16, this lane's 16 elements) from a row ptr.
// Lane layout: row = lane&15, K = {hi*16 .. +15} (pa already
// offset by hi*16); two contiguous 16B chunks at +0 and +8 elements.
__device__ __forceinline__ v16bf load_afrag(const unsigned short* pa) {
    ushort8 lo = *(const ushort8*)(pa);
    ushort8 hi = *(const ushort8*)(pa + 8);
    v16us u = __builtin_shufflevector(lo, hi,
                  0, 1, 2, 3, 4, 5, 6, 7, 8, 9, 10, 11, 12, 13, 14, 15);
    return __builtin_bit_cast(v16bf, u);
}

// Load one B-fragment (32x16 bf16) from a transposed-weight row ptr.
// Lane layout: col = lane&15, K = {hi*16 .. +15}; single contiguous 32B load.
__device__ __forceinline__ v16bf load_bfrag(const unsigned short* pb) {
    return __builtin_bit_cast(v16bf, *(const v16us*)(pb));
}

// ---------------------------------------------------------------------------
// Conversion kernels
// ---------------------------------------------------------------------------
__global__ void conv_bf16_kernel(const float* __restrict__ src,
                                 unsigned short* __restrict__ dst, int n) {
    int i = blockIdx.x * blockDim.x + threadIdx.x;
    if (i < n) dst[i] = f2bf(src[i]);
}

// W (K x N, row-major) -> WT (N x K, row-major) in bf16
__global__ void conv_wT_kernel(const float* __restrict__ W,
                               unsigned short* __restrict__ WT) {
    int i = blockIdx.x * blockDim.x + threadIdx.x;
    if (i < Dd * Dd) {
        int n = i >> 8;        // / 256
        int k = i & 255;
        WT[i] = f2bf(W[(size_t)k * Dd + n]);
    }
}

// ---------------------------------------------------------------------------
// Adjacency bitmask: Nn rows x 64 words (32 bits/word). Diagonal pre-set.
// ---------------------------------------------------------------------------
__global__ void mask_init_kernel(unsigned* __restrict__ mask) {
    int w = blockIdx.x * blockDim.x + threadIdx.x;
    if (w >= Nn * 64) return;
    int row  = w >> 6;
    int wcol = w & 63;
    mask[w] = ((row >> 5) == wcol) ? (1u << (row & 31)) : 0u;
}

__global__ void mask_edges_kernel(const int* __restrict__ ei, int E,
                                  unsigned* __restrict__ mask) {
    int e = blockIdx.x * blockDim.x + threadIdx.x;
    if (e >= E) return;
    int r = ei[e];        // edge_index[0][e]  (row attends to col)
    int c = ei[E + e];    // edge_index[1][e]
    atomicOr(&mask[(size_t)r * 64 + (c >> 5)], 1u << (c & 31));
}

// ---------------------------------------------------------------------------
// bf16 GEMM, fp32 accumulate:  Out(MxDd) = A(MxDd) * W(DdxDd) + bias
// One wave computes a 32x32 output tile: 2 A-frags x 2 B-frags per k-step
// -> 4 independent v_wmma_f32_16x16x32_bf16 accumulator chains, ~16 flops/B.
// ---------------------------------------------------------------------------
__global__ void gemm_bf16_kernel(const unsigned short* __restrict__ A,
                                 const unsigned short* __restrict__ WT,
                                 const float* __restrict__ bias,
                                 float* __restrict__ Out, int M) {
    const int tid  = threadIdx.x;
    const int lane = tid & 31;
    const int wv   = tid >> 5;
    const int gw   = blockIdx.x * (blockDim.x >> 5) + wv;

    const int tiles_n = Dd / 32;                 // 8 groups of two 16-col tiles
    const int m0 = (gw / tiles_n) * 32;
    const int n0 = (gw % tiles_n) * 32;
    if (m0 >= M) return;

    const int r  = lane & 15;                    // A: row / B: column
    const int hi = lane >> 4;                    // half-wave selector

    const unsigned short* pa0 = A  + (size_t)(m0 + r)      * Dd + hi * 16;
    const unsigned short* pa1 = A  + (size_t)(m0 + 16 + r) * Dd + hi * 16;
    const unsigned short* pb0 = WT + (size_t)(n0 + r)      * Dd + hi * 16;
    const unsigned short* pb1 = WT + (size_t)(n0 + 16 + r) * Dd + hi * 16;

    v8f acc00 = {}, acc01 = {}, acc10 = {}, acc11 = {};
    for (int k0 = 0; k0 < Dd; k0 += 32) {
        v16bf a0 = load_afrag(pa0 + k0);
        v16bf a1 = load_afrag(pa1 + k0);
        v16bf b0 = load_bfrag(pb0 + k0);
        v16bf b1 = load_bfrag(pb1 + k0);
        acc00 = __builtin_amdgcn_wmma_f32_16x16x32_bf16(
                    false, a0, false, b0, (short)0, acc00, false, false);
        acc01 = __builtin_amdgcn_wmma_f32_16x16x32_bf16(
                    false, a0, false, b1, (short)0, acc01, false, false);
        acc10 = __builtin_amdgcn_wmma_f32_16x16x32_bf16(
                    false, a1, false, b0, (short)0, acc10, false, false);
        acc11 = __builtin_amdgcn_wmma_f32_16x16x32_bf16(
                    false, a1, false, b1, (short)0, acc11, false, false);
    }

    // C/D layout: VGPR v -> row (tile_m + hi*8 + v), col (tile_n + (lane&15))
    const int   na = n0 + r;
    const int   nb = n0 + 16 + r;
    const float bva = bias[na];
    const float bvb = bias[nb];
#pragma unroll
    for (int v = 0; v < 8; ++v) {
        const int ma = m0 + hi * 8 + v;
        const int mb = m0 + 16 + hi * 8 + v;
        Out[(size_t)ma * Dd + na] = acc00[v] + bva;
        Out[(size_t)ma * Dd + nb] = acc01[v] + bvb;
        Out[(size_t)mb * Dd + na] = acc10[v] + bva;
        Out[(size_t)mb * Dd + nb] = acc11[v] + bvb;
    }
}

// ---------------------------------------------------------------------------
// Sparse masked attention, one wave per (b, h, row). Online softmax over the
// ~33 set bits of the row's adjacency mask; lane-parallel over DH=32.
// Writes attended output directly as bf16 (input to the O-projection GEMM).
// ---------------------------------------------------------------------------
__global__ void attn_sparse_kernel(const float* __restrict__ Q,
                                   const float* __restrict__ K,
                                   const float* __restrict__ V,
                                   const unsigned* __restrict__ mask,
                                   unsigned short* __restrict__ Abf) {
    const int tid  = threadIdx.x;
    const int lane = tid & 31;
    const int wv   = tid >> 5;
    const int gw   = blockIdx.x * (blockDim.x >> 5) + wv;  // [0, B*H*N)

    const int b   = gw / (Hh * Nn);
    const int rem = gw - b * Hh * Nn;
    const int h   = rem / Nn;
    const int n   = rem - h * Nn;

    const size_t rowQ = ((size_t)(b * Nn + n)) * Dd + h * DHh;
    const float  q    = Q[rowQ + lane];
    const float  scale = 0.17677669529663687f;   // 1/sqrt(32)

    float m = -INFINITY, s = 0.f, o = 0.f;
    const unsigned* mrow = mask + (size_t)n * 64;

    for (int w = 0; w < 64; ++w) {
        unsigned bits = mrow[w];
        while (bits) {
            const int c = (w << 5) + __builtin_ctz(bits);
            bits &= bits - 1;
            const size_t rowK = ((size_t)(b * Nn + c)) * Dd + h * DHh;
            float d = q * K[rowK + lane];
#pragma unroll
            for (int off = 16; off; off >>= 1) d += __shfl_xor(d, off, 32);
            d *= scale;
            const float mn   = fmaxf(m, d);
            const float corr = __expf(m - mn);   // 0 on first hit (m = -inf)
            const float p    = __expf(d - mn);
            o = o * corr + p * V[rowK + lane];
            s = s * corr + p;
            m = mn;
        }
    }
    // diagonal always set -> s > 0
    Abf[rowQ + lane] = f2bf(o / s);
}

// ---------------------------------------------------------------------------
// Residual + LayerNorm: one 256-thread block per row.
// ---------------------------------------------------------------------------
__global__ void resid_ln_kernel(const float* __restrict__ OutP,
                                const float* __restrict__ X,
                                const float* __restrict__ gamma,
                                const float* __restrict__ beta,
                                float* __restrict__ out) {
    const int row = blockIdx.x;
    const int tid = threadIdx.x;
    const size_t idx = (size_t)row * Dd + tid;

    const float y = OutP[idx] + X[idx];
    float s = y, s2 = y * y;
#pragma unroll
    for (int off = 16; off; off >>= 1) {
        s  += __shfl_xor(s,  off, 32);
        s2 += __shfl_xor(s2, off, 32);
    }
    __shared__ float sh[8], sh2[8];
    const int lane = tid & 31, wv = tid >> 5;
    if (lane == 0) { sh[wv] = s; sh2[wv] = s2; }
    __syncthreads();
    float tot = 0.f, tot2 = 0.f;
#pragma unroll
    for (int j = 0; j < 8; ++j) { tot += sh[j]; tot2 += sh2[j]; }
    const float mu  = tot * (1.0f / Dd);
    const float var = tot2 * (1.0f / Dd) - mu * mu;
    out[idx] = (y - mu) * rsqrtf(var + EPSc) * gamma[tid] + beta[tid];
}

// ---------------------------------------------------------------------------
// Host-side launch
// ---------------------------------------------------------------------------
extern "C" void kernel_launch(void* const* d_in, const int* in_sizes, int n_in,
                              void* d_out, int out_size, void* d_ws, size_t ws_size,
                              hipStream_t stream) {
    const float* x     = (const float*)d_in[0];
    const int*   ei    = (const int*)  d_in[1];
    const float* Wq    = (const float*)d_in[2];
    const float* bq    = (const float*)d_in[3];
    const float* Wk    = (const float*)d_in[4];
    const float* bk    = (const float*)d_in[5];
    const float* Wv    = (const float*)d_in[6];
    const float* bv    = (const float*)d_in[7];
    const float* Wo    = (const float*)d_in[8];
    const float* bo    = (const float*)d_in[9];
    const float* gamma = (const float*)d_in[10];
    const float* beta  = (const float*)d_in[11];
    float* out = (float*)d_out;

    const int E = in_sizes[1] / 2;
    const int M = Bb * Nn;                 // 4096 rows

    // Workspace carve-up (all offsets 256B aligned)
    char*  w   = (char*)d_ws;
    size_t off = 0;
    auto alloc = [&](size_t bytes) -> void* {
        void* p = w + off;
        off += (bytes + 255) & ~(size_t)255;
        return p;
    };
    unsigned short* Xbf  = (unsigned short*)alloc((size_t)M * Dd * 2);
    unsigned short* WqT  = (unsigned short*)alloc((size_t)Dd * Dd * 2);
    unsigned short* WkT  = (unsigned short*)alloc((size_t)Dd * Dd * 2);
    unsigned short* WvT  = (unsigned short*)alloc((size_t)Dd * Dd * 2);
    unsigned short* WoT  = (unsigned short*)alloc((size_t)Dd * Dd * 2);
    float*          Qf   = (float*)alloc((size_t)M * Dd * 4);
    float*          Kf   = (float*)alloc((size_t)M * Dd * 4);
    float*          Vf   = (float*)alloc((size_t)M * Dd * 4);
    unsigned*       mask = (unsigned*)alloc((size_t)Nn * 64 * 4);
    unsigned short* Abf  = (unsigned short*)alloc((size_t)M * Dd * 2);
    float*          OutP = (float*)alloc((size_t)M * Dd * 4);
    (void)ws_size; (void)n_in; (void)out_size;

    // 1) bf16 conversions
    conv_bf16_kernel<<<(M * Dd + 255) / 256, 256, 0, stream>>>(x, Xbf, M * Dd);
    conv_wT_kernel<<<(Dd * Dd + 255) / 256, 256, 0, stream>>>(Wq, WqT);
    conv_wT_kernel<<<(Dd * Dd + 255) / 256, 256, 0, stream>>>(Wk, WkT);
    conv_wT_kernel<<<(Dd * Dd + 255) / 256, 256, 0, stream>>>(Wv, WvT);
    conv_wT_kernel<<<(Dd * Dd + 255) / 256, 256, 0, stream>>>(Wo, WoT);

    // 2) adjacency bitmask (diag + edges, duplicates collapse like the ref mask)
    mask_init_kernel<<<(Nn * 64 + 255) / 256, 256, 0, stream>>>(mask);
    mask_edges_kernel<<<(E + 255) / 256, 256, 0, stream>>>(ei, E, mask);

    // 3) Q/K/V projections — WMMA GEMMs (one wave per 32x32 tile, 4 waves/block)
    const int ntiles  = (M / 32) * (Dd / 32);   // 1024
    const int gblocks = ntiles / 4;             // 256 blocks of 128 threads
    gemm_bf16_kernel<<<gblocks, 128, 0, stream>>>(Xbf, WqT, bq, Qf, M);
    gemm_bf16_kernel<<<gblocks, 128, 0, stream>>>(Xbf, WkT, bk, Kf, M);
    gemm_bf16_kernel<<<gblocks, 128, 0, stream>>>(Xbf, WvT, bv, Vf, M);

    // 4) sparse masked attention (one wave per (b,h,row))
    const int awaves = Bb * Hh * Nn;            // 32768
    attn_sparse_kernel<<<awaves / 4, 128, 0, stream>>>(Qf, Kf, Vf, mask, Abf);

    // 5) output projection — WMMA GEMM
    gemm_bf16_kernel<<<gblocks, 128, 0, stream>>>(Abf, WoT, bo, OutP, M);

    // 6) residual + layernorm -> d_out (fp32)
    resid_ln_kernel<<<M, Dd, 0, stream>>>(OutP, x, gamma, beta, out);
}
